// RegionProposalNetwork_78666620994001
// MI455X (gfx1250) — matrix operand order repaired and checked by reading, not compiled, not run on hardware
//
#include <hip/hip_runtime.h>
#include <stdint.h>
#include <stddef.h>

// ---------------------------------------------------------------------------
// RPN forward for MI455X (gfx1250): bf16 WMMA implicit-GEMM conv3x3 + fused
// 1x1 heads + anchor/roi epilogue.  wave32, v_wmma_f32_16x16x32_bf16,
// async global->LDS tile staging (ASYNCcnt) when the toolchain exposes it.
// ---------------------------------------------------------------------------

typedef __attribute__((ext_vector_type(16))) __bf16 v16bf;
typedef __attribute__((ext_vector_type(8)))  float  v8f;
typedef __attribute__((ext_vector_type(4)))  int    v4i;

#define NPIX   16384        // 128*128
#define CDIM   512
#define KDIM   4608         // 9*512
#define NOUT   512
#define N2     80           // 48 loc + 24 score + 8 pad
#define LOC_SZ 786432       // 16384*12*4
#define CLS_SZ 393216       // 16384*12*2

// ---- CDNA5 async global->LDS copy (guarded; fallback = VGPR staging) ----
#if __has_builtin(__builtin_amdgcn_global_load_async_to_lds_b128) && \
    __has_builtin(__builtin_amdgcn_s_wait_asynccnt)
#define USE_ASYNC 1
#warning "CDNA5: global_load_async_to_lds_b128 path ENABLED"
#else
#define USE_ASYNC 0
#warning "CDNA5: async-to-LDS builtins not available, using VGPR staging fallback"
#endif

// builtin signature (from compiler diagnostic): param0 = int4 in global AS,
// param1 = LDS-side pointer; trailing (imm offset, cpol).
typedef __attribute__((address_space(1))) v4i* gvec_ptr;
typedef __attribute__((address_space(3))) v4i* lvec_ptr;

__device__ __forceinline__ void cp16_async(const void* g, void* l) {
#if USE_ASYNC
  __builtin_amdgcn_global_load_async_to_lds_b128((gvec_ptr)g, (lvec_ptr)l, 0, 0);
#else
  *(uint4*)l = *(const uint4*)g;
#endif
}
__device__ __forceinline__ void cp_async_wait() {
#if USE_ASYNC
  __builtin_amdgcn_s_wait_asynccnt(0);
#endif
}

// float -> bf16 (round to nearest even)
__device__ __forceinline__ uint16_t f2bf(float f) {
  uint32_t u = __float_as_uint(f);
  uint32_t r = u + 0x7FFFu + ((u >> 16) & 1u);
  return (uint16_t)(r >> 16);
}

union AU { v16bf v; uint16_t u[16]; };

// -------------------- packing kernels --------------------
__global__ void k_pack_x(const float* __restrict__ x, uint16_t* __restrict__ xh) {
  int idx = blockIdx.x * 256 + threadIdx.x;           // over NPIX*CDIM
  int p = idx >> 9;
  int c = idx & 511;
  xh[idx] = f2bf(x[(size_t)c * NPIX + p]);            // NCHW -> NHWC bf16
}

__global__ void k_pack_w1(const float* __restrict__ w, uint16_t* __restrict__ wN) {
  int idx = blockIdx.x * 256 + threadIdx.x;           // over NOUT*KDIM
  int n = idx / KDIM;
  int k = idx - n * KDIM;
  int tap = k >> 9;                                   // (ky*3+kx)
  int c   = k & 511;
  wN[idx] = f2bf(w[((size_t)n * CDIM + c) * 9 + tap]);
}

__global__ void k_pack_w2(const float* __restrict__ locw,
                          const float* __restrict__ scow,
                          uint16_t* __restrict__ w2) {
  int idx = blockIdx.x * 256 + threadIdx.x;           // over N2*CDIM
  int j = idx >> 9;
  int c = idx & 511;
  float v = 0.f;
  if (j < 48)      v = locw[(size_t)j * CDIM + c];
  else if (j < 72) v = scow[(size_t)(j - 48) * CDIM + c];
  w2[idx] = f2bf(v);
}

// -------------------- conv 3x3 as implicit GEMM --------------------
// block tile: M=128 (pixels) x N=64 (out channels); 8 waves, each 32x32.
// K-loop: tap (9) outer x cb (512/32) inner -> halo math hoisted per tap.
#define MT 128
#define NT 64
#define KB 32
#define APITCH 40   // bf16 elems per LDS row (80B: 16B-aligned + bank spread)
#define BPITCH 40

__launch_bounds__(256)
__global__ void k_conv3x3(const uint16_t* __restrict__ xh,
                          const uint16_t* __restrict__ wN,
                          const float* __restrict__ bias,
                          uint16_t* __restrict__ midh) {
  __shared__ uint16_t As[MT * APITCH];   // 10240 B
  __shared__ uint16_t Bs[NT * BPITCH];   //  5120 B

  const int tid  = threadIdx.x;
  const int lane = tid & 31;
  const int wv   = tid >> 5;
  const int wm   = wv & 3;               // 0..3  -> M sub-block
  const int wn   = wv >> 2;              // 0..1  -> N sub-block
  const int l15  = lane & 15;
  const int hi   = lane >> 4;            // lane half
  const int p0   = blockIdx.y * MT;
  const int n0   = blockIdx.x * NT;

  const int akoff = hi ? 16 : 0;         // A frag K offset (contiguous k = 16*hi + e)
  const int bkoff = hi ? 16 : 0;         // B frag K offset

  // A-tile loader role: row i = tid>>1, 32B half
  const int ai    = tid >> 1;
  const int ahalf = (tid & 1) * 16;      // element offset
  uint16_t* adst = As + ai * APITCH + ahalf;
  // B-tile loader role: col j = tid>>2, 16B quarter
  const int bj   = tid >> 2;
  const int bq   = (tid & 3) * 8;        // element offset
  uint16_t* bdst = Bs + bj * BPITCH + bq;
  const uint16_t* bsrc_col = wN + (size_t)(n0 + bj) * KDIM + bq;

  // frag source pointers (fixed)
  const uint16_t* afrag0 = As + (wm * 32 + l15) * APITCH + akoff;
  const uint16_t* bfrag0 = Bs + (wn * 32 + l15) * BPITCH + bkoff;

  v8f acc[2][2] = {};

#pragma unroll 1
  for (int tap = 0; tap < 9; ++tap) {
    const int dy = tap / 3 - 1;          // hoisted out of K inner loop
    const int dx = tap % 3 - 1;
    const int p = p0 + ai;
    const int h = (p >> 7) + dy;
    const int w = (p & 127) + dx;
    const bool valid = ((unsigned)h < 128u) & ((unsigned)w < 128u);
    const uint16_t* asrc =
        xh + (size_t)((h << 7) + w) * CDIM + ahalf;  // + cb in loop
    if (!valid) {               // halo row: zero once (two writers per row, one per 32B half)
      uint4 z = {0, 0, 0, 0};
      ((uint4*)adst)[0] = z;
      ((uint4*)adst)[1] = z;
    }
    const uint16_t* bsrc = bsrc_col + tap * 512;

#pragma unroll 1
    for (int cb = 0; cb < 512; cb += KB) {
      // ---- stage tiles (async -> LDS on CDNA5) ----
      if (valid) {
        cp16_async(asrc + cb, adst);
        cp16_async(asrc + cb + 8, adst + 8);
      }
      cp16_async(bsrc, bdst);
      if (cb + KB < 512)                 // CDNA5 global_prefetch_b8
        __builtin_prefetch(bsrc + KB, 0, 1);
      bsrc += KB;
      cp_async_wait();
      __syncthreads();

      // ---- fragments + WMMA ----
      AU af[2], bf[2];
#pragma unroll
      for (int mi = 0; mi < 2; ++mi) {
        const uint16_t* s = afrag0 + mi * 16 * APITCH;
#pragma unroll
        for (int e = 0; e < 16; ++e) af[mi].u[e] = s[e];
      }
#pragma unroll
      for (int ni = 0; ni < 2; ++ni) {
        const uint16_t* s = bfrag0 + ni * 16 * BPITCH;
#pragma unroll
        for (int e = 0; e < 16; ++e) bf[ni].u[e] = s[e];
      }
#pragma unroll
      for (int mi = 0; mi < 2; ++mi)
#pragma unroll
        for (int ni = 0; ni < 2; ++ni)
          acc[mi][ni] = __builtin_amdgcn_wmma_f32_16x16x32_bf16(
              false, af[mi].v, false, bf[ni].v, (short)0, acc[mi][ni], false, false);
      __syncthreads();
    }
  }

  // ---- epilogue: bias + ReLU -> bf16 NHWC mid ----
#pragma unroll
  for (int mi = 0; mi < 2; ++mi)
#pragma unroll
    for (int ni = 0; ni < 2; ++ni) {
      const int n = n0 + wn * 32 + ni * 16 + l15;
      const float b = bias[n];
#pragma unroll
      for (int r = 0; r < 8; ++r) {
        const int m = p0 + wm * 32 + mi * 16 + r + 8 * hi;
        float v = acc[mi][ni][r] + b;
        v = v > 0.f ? v : 0.f;
        midh[(size_t)m * NOUT + n] = f2bf(v);
      }
    }
}

// -------------------- 1x1 heads + anchors/roi --------------------
// GEMM 16384 x 80 x 512.  Block: 128 rows (8 waves x 16), all 80 cols.
#define KH 256
#define W2PITCH (KH + 8)   // 264 elems (528B) -> bank-conflict-free frag reads

__launch_bounds__(256)
__global__ void k_head(const uint16_t* __restrict__ midh,
                       const uint16_t* __restrict__ w2,
                       const float* __restrict__ locb,
                       const float* __restrict__ scob,
                       float* __restrict__ out) {
  __shared__ uint16_t Ws[N2 * W2PITCH];   // 42240 B

  const int tid  = threadIdx.x;
  const int lane = tid & 31;
  const int wv   = tid >> 5;
  const int l15  = lane & 15;
  const int hi   = lane >> 4;
  const int m0   = blockIdx.x * 128 + wv * 16;

  const int akoff = hi ? 16 : 0;
  const int bkoff = hi ? 16 : 0;
  const int arow  = m0 + l15;

  v8f acc[5] = {};

#pragma unroll 1
  for (int kh = 0; kh < 2; ++kh) {
    __syncthreads();
    // stage half of w2: 80 rows x 256 bf16 (512B = 32 lanes x 16B), async
    for (int rr = wv; rr < N2; rr += 8) {
      cp16_async(w2 + (size_t)rr * CDIM + kh * KH + lane * 8,
                 Ws + rr * W2PITCH + lane * 8);
    }
    cp_async_wait();
    __syncthreads();

    for (int k0 = 0; k0 < KH; k0 += KB) {
      AU a;
      const uint16_t* s = midh + (size_t)arow * CDIM + kh * KH + k0 + akoff;
#pragma unroll
      for (int e = 0; e < 16; ++e) a.u[e] = s[e];
#pragma unroll
      for (int nt = 0; nt < 5; ++nt) {
        AU b;
        const uint16_t* t = Ws + (nt * 16 + l15) * W2PITCH + k0 + bkoff;
#pragma unroll
        for (int e = 0; e < 16; ++e) b.u[e] = t[e];
        acc[nt] = __builtin_amdgcn_wmma_f32_16x16x32_bf16(
            false, a.v, false, b.v, (short)0, acc[nt], false, false);
      }
    }
  }

  // anchors: cx=gx, cy=gy, aw=round(ws/2), ah=round(hs/2); a = ri*4+si
  const float AWt[12] = {45.f, 91.f, 181.f, 362.f, 32.f, 64.f, 128.f, 256.f,
                         23.f, 45.f, 91.f, 181.f};
  const float AHt[12] = {23.f, 45.f, 91.f, 181.f, 32.f, 64.f, 128.f, 256.f,
                         45.f, 91.f, 181.f, 362.f};
  float* d_loc = out;
  float* d_cls = out + LOC_SZ;
  float* d_roi = out + LOC_SZ + CLS_SZ;

#pragma unroll
  for (int nt = 0; nt < 5; ++nt) {
    const int j = nt * 16 + l15;       // 0..79
    if (j >= 72) continue;
    const float bj = (j < 48) ? locb[j] : scob[j - 48];
#pragma unroll
    for (int r = 0; r < 8; ++r) {
      const int m = m0 + r + 8 * hi;
      const float v = acc[nt][r] + bj;
      if (j < 48) {
        const int a = j >> 2, q = j & 3;
        const int row = m * 12 + a;
        d_loc[(size_t)row * 4 + q] = v;
        // anchor grid pairs row-for-row with outputs (x-major per reference)
        const int xi = m >> 7, yi = m & 127;
        float rv;
        if (q == 0)      rv = v * AWt[a] + (float)(xi * 16);
        else if (q == 1) rv = v * AHt[a] + (float)(yi * 16);
        else if (q == 2) rv = expf(v) * AWt[a];
        else             rv = expf(v) * AHt[a];
        d_roi[(size_t)row * 4 + q] = rv;
      } else {
        const int c2 = j - 48;
        d_cls[(size_t)(m * 12 + (c2 >> 1)) * 2 + (c2 & 1)] = v;
      }
    }
  }
}

// -------------------- launch --------------------
extern "C" void kernel_launch(void* const* d_in, const int* in_sizes, int n_in,
                              void* d_out, int out_size, void* d_ws, size_t ws_size,
                              hipStream_t stream) {
  const float* x      = (const float*)d_in[0];
  const float* conv1w = (const float*)d_in[1];
  const float* conv1b = (const float*)d_in[2];
  const float* scow   = (const float*)d_in[3];
  const float* scob   = (const float*)d_in[4];
  const float* locw   = (const float*)d_in[5];
  const float* locb   = (const float*)d_in[6];

  uint8_t* ws = (uint8_t*)d_ws;
  uint16_t* xh   = (uint16_t*)(ws);                               // 16.78 MB
  uint16_t* midh = (uint16_t*)(ws + (size_t)16777216);            // 16.78 MB
  uint16_t* wN   = (uint16_t*)(ws + (size_t)2 * 16777216);        //  4.72 MB
  uint16_t* w2   = (uint16_t*)(ws + (size_t)2 * 16777216 + 4718592);

  k_pack_x <<<(NPIX * CDIM) / 256, 256, 0, stream>>>(x, xh);
  k_pack_w1<<<(NOUT * KDIM) / 256, 256, 0, stream>>>(conv1w, wN);
  k_pack_w2<<<(N2 * CDIM) / 256, 256, 0, stream>>>(locw, scow, w2);
  k_conv3x3<<<dim3(NOUT / NT, NPIX / MT), 256, 0, stream>>>(xh, wN, conv1b, midh);
  k_head   <<<NPIX / 128, 256, 0, stream>>>(midh, w2, locb, scob, (float*)d_out);
}
